// SelfAttentionHybrid_65481071407869
// MI455X (gfx1250) — matrix operand order repaired and checked by reading, not compiled, not run on hardware
//
#include <hip/hip_runtime.h>
#include <stdint.h>

#define D 768
#define NROWS 8192
#define DBLK 256            // d-columns handled per pass (blockIdx.y)
#define NT (DBLK / 16)      // 16 C tiles per wave
#define CHUNK 32            // K of one f16 WMMA
#define NCHUNK (NROWS / CHUNK)
#define WAVES 8
#define BLOCK (WAVES * 32)  // 256 threads, 128 output rows per block
#define XSTR 40             // LDS X row stride in halves (80 B, padded)
#define XROWB (XSTR * 2)    // 80 bytes
#define BUFB (DBLK * XROWB) // 20480 bytes per LDS chunk buffer

typedef __attribute__((ext_vector_type(16))) _Float16 v16h;
typedef __attribute__((ext_vector_type(8)))  _Float16 v8h;
typedef __attribute__((ext_vector_type(8)))  float    v8f;
typedef __attribute__((ext_vector_type(4)))  float    v4f;

// ---- CDNA5 async copy: global -> LDS, 16B per lane, tracked by ASYNCcnt ----
__device__ __forceinline__ void async_ld_b128(unsigned ldsOff, const void* base,
                                              unsigned gOff) {
  asm volatile("global_load_async_to_lds_b128 %0, %1, %2"
               :
               : "v"(ldsOff), "v"(gOff), "s"(base)
               : "memory");
}
__device__ __forceinline__ void wait_async0() {
  asm volatile("s_wait_asynccnt 0" ::: "memory");
}

// ---------------- Pass 1: q/k rank-3 projections + squared norms ------------
__global__ void __launch_bounds__(BLOCK)
qk_project_kernel(const float* __restrict__ rp, const float* __restrict__ ep,
                  const float* __restrict__ X, v4f* __restrict__ qbuf,
                  v4f* __restrict__ kbuf) {
  __shared__ float sR[3 * D];
  __shared__ float sE[3 * D];
  const int tid = threadIdx.x;
  for (int i = tid; i < 3 * D; i += BLOCK) { sR[i] = rp[i]; sE[i] = ep[i]; }
  __syncthreads();

  const int wv = tid >> 5, lane = tid & 31;
  const int row = blockIdx.x * WAVES + wv;
  const float* xr = X + (size_t)row * D;

  float q0 = 0, q1 = 0, q2 = 0, k0 = 0, k1 = 0, k2 = 0;
  for (int d = lane; d < D; d += 32) {
    const float x = xr[d];
    q0 += x * sR[3 * d + 0]; q1 += x * sR[3 * d + 1]; q2 += x * sR[3 * d + 2];
    k0 += x * sE[3 * d + 0]; k1 += x * sE[3 * d + 1]; k2 += x * sE[3 * d + 2];
  }
#pragma unroll
  for (int off = 16; off > 0; off >>= 1) {
    q0 += __shfl_down(q0, off, 32); q1 += __shfl_down(q1, off, 32);
    q2 += __shfl_down(q2, off, 32); k0 += __shfl_down(k0, off, 32);
    k1 += __shfl_down(k1, off, 32); k2 += __shfl_down(k2, off, 32);
  }
  if (lane == 0) {
    v4f q = {q0, q1, q2, q0 * q0 + q1 * q1 + q2 * q2};
    v4f k = {k0, k1, k2, k0 * k0 + k1 * k1 + k2 * k2};
    qbuf[row] = q;
    kbuf[row] = k;
  }
}

// -------- Pass 0b: X (f32, row-major n x d) -> Xt (f16, d-major) ------------
#define TT 64
__global__ void __launch_bounds__(256)
transpose_f16_kernel(const float* __restrict__ X, _Float16* __restrict__ Xt) {
  __shared__ _Float16 tile[TT * 72];  // 72-half stride: 144 B, 16B-aligned rows
  const int t = threadIdx.x;
  const int n0 = blockIdx.x * TT;
  const int d0 = blockIdx.y * TT;
  const int dl4 = (t & 15) * 4;
  const int nl0 = t >> 4;
#pragma unroll
  for (int it = 0; it < 4; ++it) {
    const int nl = nl0 + 16 * it;
    const v4f x4 = *(const v4f*)(X + (size_t)(n0 + nl) * D + d0 + dl4);
#pragma unroll
    for (int i = 0; i < 4; ++i) tile[(dl4 + i) * 72 + nl] = (_Float16)x4[i];
  }
  __syncthreads();
#pragma unroll
  for (int it = 0; it < 2; ++it) {
    const int s = t + 256 * it;          // 0..511 segments of 8 halves
    const int dl = s >> 3, ns = (s & 7) * 8;
    const v8h v = *(const v8h*)&tile[dl * 72 + ns];
    *(v8h*)(Xt + (size_t)(d0 + dl) * NROWS + n0 + ns) = v;
  }
}

// ------------------ B-tile fetch from transposed LDS buffer -----------------
__device__ __forceinline__ v16h load_B(const _Float16* sXb, int t, int ml,
                                       int hi) {
  const _Float16* bp = sXb + (size_t)(t * 16 + ml) * XSTR + (hi << 4);
  const v8h blo = *(const v8h*)(bp);
  const v8h bhi = *(const v8h*)(bp + 8);
  return __builtin_shufflevector(blo, bhi, 0, 1, 2, 3, 4, 5, 6, 7, 8, 9, 10,
                                 11, 12, 13, 14, 15);
}

// ------- shared compute core: A build + 16 WMMAs, B pipelined 2 deep --------
__device__ __forceinline__ void
wmma_step(const v4f& qv, const v4f* sKb, const _Float16* sXb, int hi, int ml,
          float (&rs)[4], v8f (&C)[NT]) {
  // issue first two B-tile loads early; the exp chain below covers them
  v16h B0 = load_B(sXb, 0, ml, hi);
  v16h B1 = load_B(sXb, 1, ml, hi);

  v16h A;
#pragma unroll
  for (int e = 0; e < 16; ++e) {
    const int K = e + (hi << 4);  // f16 A-matrix K mapping (16x32), contiguous k = 16*hi + e
    const v4f kv = sKb[K];
    float d2 = qv[3] + kv[3] -
               2.0f * (qv[0] * kv[0] + qv[1] * kv[1] + qv[2] * kv[2]);
    d2 = fmaxf(d2, 0.0f);
    const float w = __expf(__expf(-d2));  // GAMMA = 1.0; logits in (0,1]
    rs[e & 3] += w;  // 4 independent partial sums: no serial VALU chain
    A[e] = (_Float16)w;
  }

#pragma unroll
  for (int t = 0; t < NT; ++t) {
    const v16h Bn = (t + 2 < NT) ? load_B(sXb, t + 2, ml, hi) : B1;
    C[t] = __builtin_amdgcn_wmma_f32_16x16x32_f16(false, A, false, B0, (short)0,
                                                  C[t], false, false);
    B0 = B1;
    B1 = Bn;
  }
}

__device__ __forceinline__ void
finish_store(const float (&rs)[4], v8f (&C)[NT], float* sSum, float* out,
             int wv, int lane, int hi, int ml, int m0, int db0) {
  sSum[wv * 32 + lane] = (rs[0] + rs[1]) + (rs[2] + rs[3]);
  __syncthreads();
  float inv[8];
#pragma unroll
  for (int r = 0; r < 8; ++r) {
    const int mloc = (hi << 3) + r;  // lane m: K-set A, lane m+16: K-set B
    inv[r] = 1.0f / (sSum[wv * 32 + mloc] + sSum[wv * 32 + 16 + mloc]);
  }
#pragma unroll
  for (int t = 0; t < NT; ++t) {
#pragma unroll
    for (int r = 0; r < 8; ++r) {
      const int m = m0 + (hi << 3) + r;  // C VGPR r -> row hi*8+r, col ml
      out[(size_t)m * D + db0 + t * 16 + ml] = C[t][r] * inv[r];
    }
  }
}

// --------- Pass 2 (primary): async double-buffered staging from Xt ----------
__global__ void __launch_bounds__(BLOCK)
attn_kernel_async(const _Float16* __restrict__ Xt, const v4f* __restrict__ qbuf,
                  const v4f* __restrict__ kbuf, float* __restrict__ out) {
  __shared__ alignas(16) _Float16 sX[2][DBLK * XSTR];  // 2 x 20 KB
  __shared__ alignas(16) v4f sK[2][CHUNK];
  __shared__ float sSum[WAVES * 32];

  const int tid = threadIdx.x;
  const int lane = tid & 31, wv = tid >> 5;
  const int hi = lane >> 4, ml = lane & 15;
  const int m0 = blockIdx.x * (WAVES * 16) + wv * 16;
  const int db0 = blockIdx.y * DBLK;

  const v4f qv = qbuf[m0 + ml];

  // per-lane async transfer geometry: 1024 16B segments / 256 threads = 4 each
  unsigned ldsOff[4], gOff[4];
#pragma unroll
  for (int j = 0; j < 4; ++j) {
    const int s = tid + BLOCK * j;       // segment id
    const int dr = s >> 2, sub = s & 3;  // LDS row (d), 16B slot within row
    ldsOff[j] = (unsigned)(uintptr_t)(&sX[0][0]) + dr * XROWB + sub * 16;
    gOff[j] = (unsigned)(((db0 + dr) * NROWS + sub * 8) * 2);
  }

  // prologue: stage chunk 0 into buffer 0
#pragma unroll
  for (int j = 0; j < 4; ++j) async_ld_b128(ldsOff[j], Xt, gOff[j]);
  if (tid < CHUNK) sK[0][tid] = kbuf[tid];
  wait_async0();
  __syncthreads();

  v8f C[NT] = {};
  float rs[4] = {0.0f, 0.0f, 0.0f, 0.0f};

  for (int c = 0; c < NCHUNK; ++c) {
    const int buf = c & 1;
    const bool more = (c + 1 < NCHUNK);
    if (more) {  // issue next chunk into the free buffer (no wait)
      const unsigned bsh = (unsigned)((buf ^ 1) * BUFB);
      const unsigned nb = (unsigned)((c + 1) * CHUNK * 2);
#pragma unroll
      for (int j = 0; j < 4; ++j) async_ld_b128(ldsOff[j] + bsh, Xt, gOff[j] + nb);
      if (tid < CHUNK) sK[buf ^ 1][tid] = kbuf[(c + 1) * CHUNK + tid];
    }
    wmma_step(qv, sK[buf], &sX[buf][0], hi, ml, rs, C);
    if (more) {
      wait_async0();    // own async transfers landed
      __syncthreads();  // whole block done reading buf & next chunk visible
    }
  }
  finish_store(rs, C, sSum, out, wv, lane, hi, ml, m0, db0);
}

// --------- Pass 2 (fallback, small ws): manual transpose staging ------------
__global__ void __launch_bounds__(BLOCK)
attn_kernel_direct(const float* __restrict__ X, const v4f* __restrict__ qbuf,
                   const v4f* __restrict__ kbuf, float* __restrict__ out) {
  __shared__ alignas(16) _Float16 sX[DBLK * XSTR];
  __shared__ alignas(16) v4f sK[CHUNK];
  __shared__ float sSum[WAVES * 32];

  const int tid = threadIdx.x;
  const int lane = tid & 31, wv = tid >> 5;
  const int hi = lane >> 4, ml = lane & 15;
  const int m0 = blockIdx.x * (WAVES * 16) + wv * 16;
  const int db0 = blockIdx.y * DBLK;

  const v4f qv = qbuf[m0 + ml];
  v8f C[NT] = {};
  float rs[4] = {0.0f, 0.0f, 0.0f, 0.0f};

  const int sdd = 4 * (tid & 63);
  const int snn = tid >> 6;

  for (int c = 0; c < NCHUNK; ++c) {
    const int n0 = c * CHUNK;
    __syncthreads();
#pragma unroll
    for (int j = 0; j < 8; ++j) {
      const int nn = snn + 4 * j;
      const v4f x4 = *(const v4f*)(X + (size_t)(n0 + nn) * D + db0 + sdd);
#pragma unroll
      for (int i = 0; i < 4; ++i)
        sX[(size_t)(sdd + i) * XSTR + nn] = (_Float16)x4[i];
    }
    if (tid < CHUNK) sK[tid] = kbuf[n0 + tid];
    if (c + 1 < NCHUNK)
      __builtin_prefetch(X + (size_t)(n0 + CHUNK + snn) * D + db0 + sdd, 0, 1);
    __syncthreads();
    wmma_step(qv, sK, sX, hi, ml, rs, C);
  }
  finish_store(rs, C, sSum, out, wv, lane, hi, ml, m0, db0);
}

extern "C" void kernel_launch(void* const* d_in, const int* in_sizes, int n_in,
                              void* d_out, int out_size, void* d_ws,
                              size_t ws_size, hipStream_t stream) {
  const float* rp = (const float*)d_in[0];  // rotation_params (3*768)
  const float* ep = (const float*)d_in[1];  // entangle_params (3*768)
  const float* X  = (const float*)d_in[2];  // inputs (8192*768)
  float* out = (float*)d_out;

  v4f* qbuf = (v4f*)d_ws;
  v4f* kbuf = qbuf + NROWS;
  _Float16* Xt = (_Float16*)(kbuf + NROWS);  // 768*8192 f16 = 12 MB
  const size_t need = (size_t)2 * NROWS * sizeof(v4f) +
                      (size_t)D * NROWS * sizeof(_Float16);

  qk_project_kernel<<<NROWS / WAVES, BLOCK, 0, stream>>>(rp, ep, X, qbuf, kbuf);

  dim3 grid(NROWS / (WAVES * 16), D / DBLK);  // (64, 3)
  if (ws_size >= need) {
    transpose_f16_kernel<<<dim3(NROWS / TT, D / TT), 256, 0, stream>>>(X, Xt);
    attn_kernel_async<<<grid, BLOCK, 0, stream>>>(Xt, qbuf, kbuf, out);
  } else {
    attn_kernel_direct<<<grid, BLOCK, 0, stream>>>(X, qbuf, kbuf, out);
  }
}
